// CodeBook_60146722013379
// MI455X (gfx1250) — compile-verified, matrix operand order checked by reading, not run on hardware
//
#include <hip/hip_runtime.h>
#include <hip/hip_bf16.h>
#include <stdint.h>

#define NROW 32768
#define KCB  8192
#define DDIM 256

typedef __bf16 bf16t;
typedef bf16t v16bf __attribute__((ext_vector_type(16)));
typedef float v8f  __attribute__((ext_vector_type(8)));

union Frag {
  v16bf v;
  uint4 q[2];
};

__device__ __forceinline__ unsigned short f2bf_bits(float f) {
  unsigned int u = __float_as_uint(f);
  u += 0x7FFFu + ((u >> 16) & 1u);   // round-to-nearest-even
  return (unsigned short)(u >> 16);
}
__device__ __forceinline__ float bf2f(unsigned short b) {
  return __uint_as_float(((unsigned int)b) << 16);
}

// ---------------------------------------------------------------------------
// Kernel 0: split codebook into bf16 hi/lo, compute ||c||^2, zero loss accum.
// ---------------------------------------------------------------------------
__global__ void vq_prep(const float* __restrict__ cb,
                        unsigned short* __restrict__ ch,
                        unsigned short* __restrict__ cl,
                        float* __restrict__ c2,
                        double* __restrict__ lossAcc) {
  const int k = blockIdx.x;
  const int d = threadIdx.x;
  const float f = cb[(size_t)k * DDIM + d];
  const unsigned short hb = f2bf_bits(f);
  ch[(size_t)k * DDIM + d] = hb;
  cl[(size_t)k * DDIM + d] = f2bf_bits(f - bf2f(hb));

  __shared__ float s[DDIM];
  s[d] = f * f;
  __syncthreads();
  for (int off = DDIM / 2; off > 0; off >>= 1) {
    if (d < off) s[d] += s[d + off];
    __syncthreads();
  }
  if (d == 0) {
    c2[k] = s[0];
    if (k == 0) *lossAcc = 0.0;
  }
}

// ---------------------------------------------------------------------------
// Kernel 1: fused GEMM(+argmin) with bf16x3 WMMA.
// Block = 256 threads = 8 waves; block owns 32 rows of x.
// wave w: row-tile (w>>2) of 16 rows, column range (w&3) of K/4 codes.
// Each wave: 128 chunks of 16 codes; per chunk 8 K-blocks x 3 wmma.
// ---------------------------------------------------------------------------
__launch_bounds__(256, 1)
__global__ void vq_argmin(const float* __restrict__ x,
                          const unsigned short* __restrict__ ch,
                          const unsigned short* __restrict__ cl,
                          const float* __restrict__ c2,
                          int* __restrict__ enc,
                          float* __restrict__ encf) {
  __shared__ unsigned short xs_hi[32][264];   // pad: 264*2B = 33*16B, bank-safe
  __shared__ unsigned short xs_lo[32][264];
  __shared__ float redv[32][4];
  __shared__ int   redi[32][4];

  const int tid = threadIdx.x;
  const int row0 = blockIdx.x * 32;

  // Stage + split x tile (32 x 256) into LDS as bf16 hi/lo.
  for (int e = tid; e < 32 * DDIM; e += 256) {
    const int r = e >> 8;
    const int dd = e & 255;
    const float f = x[(size_t)(row0 + r) * DDIM + dd];
    const unsigned short hb = f2bf_bits(f);
    xs_hi[r][dd] = hb;
    xs_lo[r][dd] = f2bf_bits(f - bf2f(hb));
  }
  __syncthreads();

  const int w = tid >> 5;      // wave id (wave32)
  const int lane = tid & 31;
  const int m = lane & 15;     // A row / B col within tile
  const int h = lane >> 4;     // half-wave
  const int rowtile = w >> 2;  // 0..1
  const int colset  = w & 3;   // 0..3

  // Hoist A fragments (hi & lo) for the whole row-tile into registers.
  // 16-bit A layout: lane l -> M = l%16; K = dbase + {8h..8h+7, 16+8h..16+8h+7}
  Frag a_hi[8], a_lo[8];
  const int arow = rowtile * 16 + m;
#pragma unroll
  for (int db = 0; db < 8; ++db) {
    const int dbase = db * 32 + 8 * h;
    a_hi[db].q[0] = *(const uint4*)&xs_hi[arow][dbase];
    a_hi[db].q[1] = *(const uint4*)&xs_hi[arow][dbase + 16];
    a_lo[db].q[0] = *(const uint4*)&xs_lo[arow][dbase];
    a_lo[db].q[1] = *(const uint4*)&xs_lo[arow][dbase + 16];
  }

  float best[8];
  int   bidx[8];
#pragma unroll
  for (int v = 0; v < 8; ++v) { best[v] = 3.402823466e38f; bidx[v] = 0; }

  for (int chunk = 0; chunk < (KCB / 4) / 16; ++chunk) {
    const int mycol = colset * (KCB / 4) + chunk * 16 + m;
    const unsigned short* chp = ch + (size_t)mycol * DDIM;
    const unsigned short* clp = cl + (size_t)mycol * DDIM;
    __builtin_prefetch(chp + 16 * DDIM, 0, 1);   // next chunk's hi row
    __builtin_prefetch(clp + 16 * DDIM, 0, 1);   // next chunk's lo row

    v8f acc = {0.f, 0.f, 0.f, 0.f, 0.f, 0.f, 0.f, 0.f};
#pragma unroll
    for (int db = 0; db < 8; ++db) {
      const int dbase = db * 32 + 8 * h;
      Frag bh, bl;
      bh.q[0] = *(const uint4*)(chp + dbase);
      bh.q[1] = *(const uint4*)(chp + dbase + 16);
      bl.q[0] = *(const uint4*)(clp + dbase);
      bl.q[1] = *(const uint4*)(clp + dbase + 16);
      // dot ~= hi*hi + hi*lo + lo*hi  (fp32 accumulate)
      acc = __builtin_amdgcn_wmma_f32_16x16x32_bf16(false, a_hi[db].v, false, bh.v,
                                                    (short)0, acc, false, false);
      acc = __builtin_amdgcn_wmma_f32_16x16x32_bf16(false, a_hi[db].v, false, bl.v,
                                                    (short)0, acc, false, false);
      acc = __builtin_amdgcn_wmma_f32_16x16x32_bf16(false, a_lo[db].v, false, bh.v,
                                                    (short)0, acc, false, false);
    }

    const float c2v = c2[mycol];
#pragma unroll
    for (int v = 0; v < 8; ++v) {
      // argmin_k(||x||^2 + ||c||^2 - 2 x.c) == argmin_k(||c||^2 - 2 x.c)
      const float score = __builtin_fmaf(-2.0f, acc[v], c2v);
      if (score < best[v]) { best[v] = score; bidx[v] = mycol; }
    }
  }

  // Reduce argmin across the 16 column-lanes (same half-wave).
#pragma unroll
  for (int off = 1; off < 16; off <<= 1) {
#pragma unroll
    for (int v = 0; v < 8; ++v) {
      const float ov = __shfl_xor(best[v], off, 32);
      const int   oi = __shfl_xor(bidx[v], off, 32);
      if (ov < best[v] || (ov == best[v] && oi < bidx[v])) {
        best[v] = ov; bidx[v] = oi;
      }
    }
  }

  // C/D layout: acc[v] lives at row M = v + 8h of the tile.
  if (m == 0) {
#pragma unroll
    for (int v = 0; v < 8; ++v) {
      const int rr = rowtile * 16 + 8 * h + v;
      redv[rr][colset] = best[v];
      redi[rr][colset] = bidx[v];
    }
  }
  __syncthreads();

  // Combine the 4 column ranges; write encoding.
  if (tid < 32) {
    float bv = redv[tid][0];
    int   bi = redi[tid][0];
#pragma unroll
    for (int cs = 1; cs < 4; ++cs) {
      const float ov = redv[tid][cs];
      const int   oi = redi[tid][cs];
      if (ov < bv || (ov == bv && oi < bi)) { bv = ov; bi = oi; }
    }
    const int grow = row0 + tid;
    enc[grow]  = bi;
    encf[grow] = (float)bi;
  }
}

// ---------------------------------------------------------------------------
// Kernel 2: gather nearest = codebook[enc], accumulate sum((x - nearest)^2).
// ---------------------------------------------------------------------------
__global__ void vq_gather_loss(const float* __restrict__ x,
                               const float* __restrict__ cb,
                               const int* __restrict__ enc,
                               float* __restrict__ nearest,
                               double* __restrict__ lossAcc) {
  const int n = blockIdx.x;
  const int d = threadIdx.x;
  const int e = enc[n];
  const float cv = cb[(size_t)e * DDIM + d];
  const float xv = x[(size_t)n * DDIM + d];
  nearest[(size_t)n * DDIM + d] = cv;
  const float diff = xv - cv;

  __shared__ float s[DDIM];
  s[d] = diff * diff;
  __syncthreads();
  for (int off = DDIM / 2; off > 0; off >>= 1) {
    if (d < off) s[d] += s[d + off];
    __syncthreads();
  }
  if (d == 0) atomicAdd(lossAcc, (double)s[0]);
}

// ---------------------------------------------------------------------------
// Kernel 3: finalize both losses (identical forward values).
// ---------------------------------------------------------------------------
__global__ void vq_finalize(const double* __restrict__ lossAcc,
                            float* __restrict__ out_losses) {
  const float mean = (float)(*lossAcc / ((double)NROW * (double)DDIM));
  out_losses[0] = mean;  // codebook_loss
  out_losses[1] = mean;  // encoder_loss
}

// ---------------------------------------------------------------------------
extern "C" void kernel_launch(void* const* d_in, const int* in_sizes, int n_in,
                              void* d_out, int out_size, void* d_ws, size_t ws_size,
                              hipStream_t stream) {
  const float* x  = (const float*)d_in[0];   // [N, D]
  const float* cb = (const float*)d_in[1];   // [K, D]
  float* out = (float*)d_out;

  // Workspace layout
  char* ws = (char*)d_ws;
  unsigned short* ch = (unsigned short*)ws;                               // 4 MB
  unsigned short* cl = (unsigned short*)(ws + (size_t)KCB * DDIM * 2);    // 4 MB
  float*  c2      = (float*)(ws + (size_t)KCB * DDIM * 4);                // 32 KB
  int*    enc     = (int*)(ws + (size_t)KCB * DDIM * 4 + KCB * 4);        // 128 KB
  double* lossAcc = (double*)(ws + (size_t)KCB * DDIM * 4 + KCB * 4 + (size_t)NROW * 4);

  // Output layout: [encoding (N floats)][loss, loss][nearest (N*D floats)]
  float* encf    = out;
  float* losses  = out + NROW;
  float* nearest = out + NROW + 2;

  vq_prep       <<<dim3(KCB),      dim3(DDIM), 0, stream>>>(cb, ch, cl, c2, lossAcc);
  vq_argmin     <<<dim3(NROW / 32), dim3(256),  0, stream>>>(x, ch, cl, c2, enc, encf);
  vq_gather_loss<<<dim3(NROW),     dim3(DDIM), 0, stream>>>(x, cb, enc, nearest, lossAcc);
  vq_finalize   <<<dim3(1),        dim3(1),    0, stream>>>(lossAcc, losses);
}
